// PPAggregator_53463752900616
// MI455X (gfx1250) — matrix operand order repaired and checked by reading, not compiled, not run on hardware
//
#include <hip/hip_runtime.h>
#include <hip/hip_bf16.h>
#include <stdint.h>

// Problem constants (match reference)
#define NN   20000
#define KK   32
#define DD   128
#define NEGV (-1e30f)

// ---- CDNA5 WMMA types (wave32) ----
typedef __attribute__((ext_vector_type(16))) __bf16     v16bf;
typedef __attribute__((ext_vector_type(8)))  float      v8f;
typedef __attribute__((ext_vector_type(4)))  unsigned int u32x4;

union Frag16 { u32x4 q[2]; v16bf v; };   // 8 dwords = 16 bf16 per lane

__device__ __forceinline__ uint32_t pk_bf16(float a, float b) {
    uint32_t ua = __float_as_uint(a); ua = (ua + 0x7FFFu + ((ua >> 16) & 1u)) >> 16;
    uint32_t ub = __float_as_uint(b); ub = (ub + 0x7FFFu + ((ub >> 16) & 1u)) >> 16;
    return (ua & 0xFFFFu) | (ub << 16);
}
__device__ __forceinline__ uint16_t f2bf(float a) {
    uint32_t u = __float_as_uint(a);
    return (uint16_t)((u + 0x7FFFu + ((u >> 16) & 1u)) >> 16);
}
__device__ __forceinline__ float bf2f(uint16_t h) {
    return __uint_as_float(((uint32_t)h) << 16);
}

// ---------------------------------------------------------------------------
// Prologue: pack W1 (128x256 f32) and W2 (128x128 f32) into bf16 B-matrix
// fragment order for v_wmma_f32_16x16x32_bf16.
// B tile (32k x 16n): lane L<16 -> col L, K=0..15 in dwords 0..7 (2/dword);
//                     lane L+16 -> col L, K=16..31.
// Layout: Wp[((ntile*NSTEP + kstep)*32 + lane)*8 + dword]
// ---------------------------------------------------------------------------
__global__ void prepack_weights(const float* __restrict__ W1,
                                const float* __restrict__ W2,
                                uint32_t* __restrict__ W1p,
                                uint32_t* __restrict__ W2p) {
    int tid = blockIdx.x * blockDim.x + threadIdx.x;
    const int NW1 = 8 * 8 * 32 * 8;   // 8 ntiles, 8 ksteps
    const int NW2 = 8 * 4 * 32 * 8;   // 8 ntiles, 4 ksteps
    if (tid < NW1) {
        int j = tid & 7, lane = (tid >> 3) & 31, s = (tid >> 8) & 7, t = tid >> 11;
        int col = t * 16 + (lane & 15);
        int f   = s * 32 + (lane >> 4) * 16 + 2 * j;
        W1p[tid] = pk_bf16(W1[col * 256 + f], W1[col * 256 + f + 1]);
    }
    int t2 = tid - NW1;
    if (t2 >= 0 && t2 < NW2) {
        int j = t2 & 7, lane = (t2 >> 3) & 31, s = (t2 >> 8) & 3, t = t2 >> 10;
        int col = t * 16 + (lane & 15);
        int f   = s * 32 + (lane >> 4) * 16 + 2 * j;
        W2p[t2] = pk_bf16(W2[col * 128 + f], W2[col * 128 + f + 1]);
    }
}

// ---------------------------------------------------------------------------
// Fused GraphRec attention aggregation. 1 block = 1 node, 128 thr = 4 waves.
// ---------------------------------------------------------------------------
__global__ __launch_bounds__(128, 1) void graphrec_fused(
    const int*   __restrict__ nodes,
    const int*   __restrict__ neigh_idx,
    const int*   __restrict__ neigh_len,
    const float* __restrict__ u2e,
    const float* __restrict__ b1,
    const float* __restrict__ b2,
    const float* __restrict__ W3,
    const float* __restrict__ b3,
    const uint32_t* __restrict__ W1p,
    const uint32_t* __restrict__ W2p,
    float* __restrict__ out) {

    __shared__ float    s_eu[KK * DD];        // 16 KB: f32 neighbor tile (aggregation)
    __shared__ uint16_t s_X [KK * 2 * DD];    // 16 KB: bf16 [e_u | self] 32x256
    __shared__ uint16_t s_H1[KK * DD];        //  8 KB: bf16 layer-1 activations
    __shared__ uint16_t s_H2[KK * DD];        //  8 KB: bf16 layer-2 activations
    __shared__ float    s_self[DD];
    __shared__ float    s_att[KK];
    __shared__ float    s_logit[KK];
    __shared__ int      s_nidx[KK];
    __shared__ int      s_len, s_node;

    const int n    = blockIdx.x;
    const int tid  = threadIdx.x;
    const int wid  = tid >> 5;      // wave id 0..3 (wave32)
    const int lane = tid & 31;
    const int l    = lane & 15;
    const int half = lane >> 4;

    if (tid < KK) s_nidx[tid] = neigh_idx[n * KK + tid];
    if (tid == 0) { s_len = neigh_len[n]; s_node = nodes[n]; }
    __syncthreads();

    // ---- gather: self row (replicated bf16 into cols 128..255 of every row)
    {
        float sv = u2e[(size_t)s_node * DD + tid];
        s_self[tid] = sv;
        uint16_t bs = f2bf(sv);
        #pragma unroll
        for (int r = 0; r < KK; ++r) s_X[r * (2 * DD) + DD + tid] = bs;
    }
    // ---- gather: 32 neighbor rows, float4 each, dual-store f32 + bf16
    #pragma unroll
    for (int it = 0; it < 8; ++it) {
        int lin = it * 128 + tid;           // 0..1023 float4 slots
        int row = lin >> 5;                 // 0..31
        int c4  = (lin & 31) << 2;          // column base
        const float4 v = *(const float4*)(u2e + (size_t)s_nidx[row] * DD + c4);
        *(float4*)(s_eu + row * DD + c4) = v;
        uint32_t* xd = (uint32_t*)(s_X + row * (2 * DD) + c4);
        xd[0] = pk_bf16(v.x, v.y);
        xd[1] = pk_bf16(v.z, v.w);
    }
    __syncthreads();

    // ---- layer 1: H1 = relu(X(32x256) * W1^T + b1), bf16 WMMA, f32 accum
    {
        v8f acc[2][2] = {};                 // [mtile][ntile]
        #pragma unroll
        for (int s = 0; s < 8; ++s) {
            Frag16 a[2], b[2];
            #pragma unroll
            for (int mt = 0; mt < 2; ++mt) {
                // contiguous A layout (matches packed B): lane<16 -> K0..15, lane>=16 -> K16..31
                int base = (mt * 16 + l) * (2 * DD) + s * 32 + half * 16;
                a[mt].q[0] = *(const u32x4*)(s_X + base);
                a[mt].q[1] = *(const u32x4*)(s_X + base + 8);
            }
            #pragma unroll
            for (int nt = 0; nt < 2; ++nt) {
                int t = wid * 2 + nt;
                const u32x4* p = (const u32x4*)(W1p + (((t * 8 + s) * 32 + lane) << 3));
                b[nt].q[0] = p[0];
                b[nt].q[1] = p[1];
            }
            #pragma unroll
            for (int mt = 0; mt < 2; ++mt)
                #pragma unroll
                for (int nt = 0; nt < 2; ++nt)
                    acc[mt][nt] = __builtin_amdgcn_wmma_f32_16x16x32_bf16(
                        false, a[mt].v, false, b[nt].v,
                        (short)0, acc[mt][nt], false, false);
        }
        #pragma unroll
        for (int nt = 0; nt < 2; ++nt) {
            int col = wid * 32 + nt * 16 + l;
            float bias = b1[col];
            #pragma unroll
            for (int mt = 0; mt < 2; ++mt)
                #pragma unroll
                for (int r = 0; r < 8; ++r) {
                    // C/D layout: VGPR r, lanes0-15 -> M=r; lanes16-31 -> M=8+r
                    float v = acc[mt][nt][r] + bias;
                    v = v > 0.f ? v : 0.f;
                    s_H1[(mt * 16 + half * 8 + r) * DD + col] = f2bf(v);
                }
        }
    }
    __syncthreads();

    // ---- layer 2: H2 = relu(H1(32x128) * W2^T + b2)
    {
        v8f acc[2][2] = {};
        #pragma unroll
        for (int s = 0; s < 4; ++s) {
            Frag16 a[2], b[2];
            #pragma unroll
            for (int mt = 0; mt < 2; ++mt) {
                int base = (mt * 16 + l) * DD + s * 32 + half * 16;
                a[mt].q[0] = *(const u32x4*)(s_H1 + base);
                a[mt].q[1] = *(const u32x4*)(s_H1 + base + 8);
            }
            #pragma unroll
            for (int nt = 0; nt < 2; ++nt) {
                int t = wid * 2 + nt;
                const u32x4* p = (const u32x4*)(W2p + (((t * 4 + s) * 32 + lane) << 3));
                b[nt].q[0] = p[0];
                b[nt].q[1] = p[1];
            }
            #pragma unroll
            for (int mt = 0; mt < 2; ++mt)
                #pragma unroll
                for (int nt = 0; nt < 2; ++nt)
                    acc[mt][nt] = __builtin_amdgcn_wmma_f32_16x16x32_bf16(
                        false, a[mt].v, false, b[nt].v,
                        (short)0, acc[mt][nt], false, false);
        }
        #pragma unroll
        for (int nt = 0; nt < 2; ++nt) {
            int col = wid * 32 + nt * 16 + l;
            float bias = b2[col];
            #pragma unroll
            for (int mt = 0; mt < 2; ++mt)
                #pragma unroll
                for (int r = 0; r < 8; ++r) {
                    float v = acc[mt][nt][r] + bias;
                    v = v > 0.f ? v : 0.f;
                    s_H2[(mt * 16 + half * 8 + r) * DD + col] = f2bf(v);
                }
        }
    }
    __syncthreads();

    // ---- logits (f32 dot with W3) + mask
    if (tid < KK) {
        float acc = 0.f;
        #pragma unroll 8
        for (int d = 0; d < DD; ++d) acc += bf2f(s_H2[tid * DD + d]) * W3[d];
        float lg = acc + b3[0];
        s_logit[tid] = (tid < s_len) ? lg : NEGV;
    }
    __syncthreads();

    // ---- softmax over K on wave 0 (threads 0..31 are exactly one wave32)
    if (tid < KK) {
        float v = s_logit[lane];
        float m = v;
        #pragma unroll
        for (int o = 16; o > 0; o >>= 1) m = fmaxf(m, __shfl_xor(m, o, 32));
        float e = __expf(v - m);
        float sum = e;
        #pragma unroll
        for (int o = 16; o > 0; o >>= 1) sum += __shfl_xor(sum, o, 32);
        s_att[lane] = (lane < s_len) ? (e / sum) : 0.f;
    }
    __syncthreads();

    // ---- f32 attention-weighted aggregation + residual (thread tid = dim d)
    float acc = 0.f;
    #pragma unroll 8
    for (int k = 0; k < KK; ++k) acc += s_att[k] * s_eu[k * DD + tid];
    float res = (s_len > 0) ? 0.5f * (acc + s_self[tid]) : s_self[tid];
    out[(size_t)n * DD + tid] = res;
}

// ---------------------------------------------------------------------------
extern "C" void kernel_launch(void* const* d_in, const int* in_sizes, int n_in,
                              void* d_out, int out_size, void* d_ws, size_t ws_size,
                              hipStream_t stream) {
    (void)in_sizes; (void)n_in; (void)out_size; (void)ws_size;
    const int*   nodes = (const int*)d_in[0];
    const int*   nidx  = (const int*)d_in[1];
    const int*   nlen  = (const int*)d_in[2];
    const float* u2e   = (const float*)d_in[3];
    const float* W1    = (const float*)d_in[4];
    const float* b1    = (const float*)d_in[5];
    const float* W2    = (const float*)d_in[6];
    const float* b2    = (const float*)d_in[7];
    const float* W3    = (const float*)d_in[8];
    const float* b3    = (const float*)d_in[9];

    uint32_t* W1p = (uint32_t*)d_ws;                 // 64 KB packed bf16 fragments
    uint32_t* W2p = W1p + 8 * 8 * 32 * 8;            // 32 KB packed bf16 fragments

    const int packN = 8 * 8 * 32 * 8 + 8 * 4 * 32 * 8;   // 24576 threads
    prepack_weights<<<(packN + 255) / 256, 256, 0, stream>>>(W1, W2, W1p, W2p);

    graphrec_fused<<<NN, 128, 0, stream>>>(nodes, nidx, nlen, u2e, b1, b2,
                                           W3, b3, W1p, W2p, (float*)d_out);
}
